// MoELayer_420906795433
// MI455X (gfx1250) — compile-verified, matrix operand order checked by reading, not run on hardware
//
#include <hip/hip_runtime.h>
#include <hip/hip_bf16.h>
#include <math.h>
#include <stdint.h>

typedef __attribute__((ext_vector_type(16))) __bf16 v16bf;
typedef __attribute__((ext_vector_type(8)))  __bf16 v8bf;
typedef __attribute__((ext_vector_type(8)))  float  v8f;

#define NDIM 1024   // d_model
#define HDIM 4096   // hidden
#define NEXP 8
#define NTOK 8192
#define MT   32     // tokens per workgroup tile (2 x 16-row WMMA sub-tiles)

// LDS: xA 32x1024 bf16 (64KB) + h 32x4096 bf16 (256KB) = 320KB exactly.
// No pad possible -> XOR-swizzle 16B chunks to avoid 16-way bank conflicts
// (row strides 2KB / 8KB are 0 mod 64 banks).
#define SMEM_A (MT * NDIM * 2)
#define SMEM_BYTES (SMEM_A + MT * HDIM * 2)

static __device__ __forceinline__ int swz_off(int row, int k, int rowHalfs) {
  int c = (k >> 3) ^ (row & 15);          // permute 8-half chunks within row
  return row * rowHalfs + (c << 3) + (k & 7);
}

static __device__ __forceinline__ v16bf cat8(v8bf lo, v8bf hi) {
  return __builtin_shufflevector(lo, hi, 0,1,2,3,4,5,6,7,8,9,10,11,12,13,14,15);
}

// Row-contiguous (global) fragment: lanes 0-15 K={0..7,16..23}, lanes 16-31
// K={8..15,24..31} per ISA 7.12.2; caller pre-offsets by kb=(lane<16?0:8).
static __device__ __forceinline__ v16bf load_frag(const __bf16* base) {
  v8bf lo = *(const v8bf*)(base);
  v8bf hi = *(const v8bf*)(base + 16);
  return cat8(lo, hi);
}

// Swizzled LDS fragment (k = k0 + kb, both 8-half chunks located separately)
static __device__ __forceinline__ v16bf load_frag_swz(const __bf16* base,
                                                      int row, int k, int rowHalfs) {
  v8bf lo = *(const v8bf*)(base + swz_off(row, k, rowHalfs));
  v8bf hi = *(const v8bf*)(base + swz_off(row, k + 16, rowHalfs));
  return cat8(lo, hi);
}

// gfx1250 async copy engine: 16B global -> LDS, tracked by ASYNCcnt.
static __device__ __forceinline__ void async_copy16(unsigned ldsByteOff, const __bf16* g) {
  unsigned long long ga = (unsigned long long)(uintptr_t)g;
  asm volatile("global_load_async_to_lds_b128 %0, %1, off"
               :: "v"(ldsByteOff), "v"(ga) : "memory");
}
static __device__ __forceinline__ void async_wait0() {
  asm volatile("s_wait_asynccnt 0" ::: "memory");
}

// ---------------------------------------------------------------------------
// 1) Gating: logits -> softmax -> top2 -> per-expert scatter lists
// ---------------------------------------------------------------------------
__global__ void moe_gate_kernel(const float* __restrict__ x,
                                const float* __restrict__ Wg,
                                const float* __restrict__ bg,
                                float* __restrict__ gate_probs,
                                float* __restrict__ hard_mask,
                                float* __restrict__ sum_probs,
                                int*   __restrict__ counts,
                                int*   __restrict__ perm,
                                float* __restrict__ cwt) {
  const int wv = threadIdx.x >> 5;
  const int lane = threadIdx.x & 31;
  const int t = blockIdx.x * 8 + wv;

  float acc[NEXP];
  #pragma unroll
  for (int e = 0; e < NEXP; ++e) acc[e] = 0.f;

  const float* xr = x + (size_t)t * NDIM;
  for (int d = lane; d < NDIM; d += 32) {
    float xv = xr[d];
    const float* wr = Wg + (size_t)d * NEXP;
    #pragma unroll
    for (int e = 0; e < NEXP; ++e) acc[e] += xv * wr[e];
  }
  #pragma unroll
  for (int off = 16; off > 0; off >>= 1) {
    #pragma unroll
    for (int e = 0; e < NEXP; ++e) acc[e] += __shfl_xor(acc[e], off);
  }

  if (lane == 0) {
    float p[NEXP];
    float m = -1e30f;
    #pragma unroll
    for (int e = 0; e < NEXP; ++e) { p[e] = acc[e] + bg[e]; m = fmaxf(m, p[e]); }
    float s = 0.f;
    #pragma unroll
    for (int e = 0; e < NEXP; ++e) { p[e] = __expf(p[e] - m); s += p[e]; }
    float inv = 1.f / s;
    #pragma unroll
    for (int e = 0; e < NEXP; ++e) {
      p[e] *= inv;
      gate_probs[(size_t)t * NEXP + e] = p[e];
      atomicAdd(&sum_probs[e], p[e]);
    }
    int b0 = 0;
    #pragma unroll
    for (int e = 1; e < NEXP; ++e) if (p[e] > p[b0]) b0 = e;
    int b1 = (b0 == 0) ? 1 : 0;
    #pragma unroll
    for (int e = 0; e < NEXP; ++e) if (e != b0 && p[e] > p[b1]) b1 = e;
    float denom = p[b0] + p[b1] + 1e-9f;
    float w0 = p[b0] / denom, w1 = p[b1] / denom;
    #pragma unroll
    for (int e = 0; e < NEXP; ++e)
      hard_mask[(size_t)t * NEXP + e] = (e == b0 || e == b1) ? 1.f : 0.f;

    int pos = atomicAdd(&counts[b0], 1);
    perm[b0 * NTOK + pos] = t;  cwt[b0 * NTOK + pos] = w0;
    pos = atomicAdd(&counts[b1], 1);
    perm[b1 * NTOK + pos] = t;  cwt[b1 * NTOK + pos] = w1;
  }
}

// ---------------------------------------------------------------------------
// 2) fp32 -> bf16 (weights transposed: W1t[e][n][k], W2t[e][d][h])
// ---------------------------------------------------------------------------
__global__ void moe_cvt_x_kernel(const float* __restrict__ src, __bf16* __restrict__ dst) {
  size_t n = (size_t)NTOK * NDIM;
  size_t stride = (size_t)gridDim.x * blockDim.x;
  for (size_t i = (size_t)blockIdx.x * blockDim.x + threadIdx.x; i < n; i += stride)
    dst[i] = (__bf16)src[i];
}

__global__ void moe_cvt_w1_kernel(const float* __restrict__ W1, __bf16* __restrict__ W1t) {
  size_t n = (size_t)NEXP * HDIM * NDIM;      // dst: e<<22 | n<<10 | k
  size_t stride = (size_t)gridDim.x * blockDim.x;
  for (size_t i = (size_t)blockIdx.x * blockDim.x + threadIdx.x; i < n; i += stride) {
    size_t e = i >> 22;
    size_t rem = i & 0x3FFFFFu;
    size_t nn = rem >> 10;
    size_t k  = rem & 1023u;
    W1t[i] = (__bf16)W1[(e << 22) + k * HDIM + nn];   // W1: [E][D][H]
  }
}

__global__ void moe_cvt_w2_kernel(const float* __restrict__ W2, __bf16* __restrict__ W2t) {
  size_t n = (size_t)NEXP * NDIM * HDIM;      // dst: e<<22 | d<<12 | h
  size_t stride = (size_t)gridDim.x * blockDim.x;
  for (size_t i = (size_t)blockIdx.x * blockDim.x + threadIdx.x; i < n; i += stride) {
    size_t e = i >> 22;
    size_t rem = i & 0x3FFFFFu;
    size_t d  = rem >> 12;
    size_t h  = rem & 4095u;
    W2t[i] = (__bf16)W2[(e << 22) + h * NDIM + d];    // W2: [E][H][D]
  }
}

// ---------------------------------------------------------------------------
// 3) aux load-balance loss
// ---------------------------------------------------------------------------
__global__ void moe_aux_kernel(const float* __restrict__ sum_probs, float* __restrict__ aux) {
  if (threadIdx.x == 0 && blockIdx.x == 0) {
    const float u = 1.f / NEXP;
    float s = 0.f;
    #pragma unroll
    for (int e = 0; e < NEXP; ++e) {
      float mp = sum_probs[e] / (float)NTOK;
      s += u * (__logf(u) - __logf(mp));
    }
    aux[0] = s / (float)NEXP;
  }
}

// ---------------------------------------------------------------------------
// 4) Fused expert FFN: 32 tokens of one expert per workgroup, 8 waves.
//    Each wave runs 2 M-sub-tiles per B fragment (halves L2 weight traffic).
//    x tile staged via global_load_async_to_lds_b128; h lives in LDS.
// ---------------------------------------------------------------------------
__global__ void __launch_bounds__(256)
moe_ffn_kernel(const __bf16* __restrict__ xb,
               const __bf16* __restrict__ W1t,
               const float*  __restrict__ b1,
               const __bf16* __restrict__ W2t,
               const float*  __restrict__ b2,
               const int*    __restrict__ counts,
               const int*    __restrict__ perm,
               const float*  __restrict__ cwt,
               float*        __restrict__ y) {
  extern __shared__ char smem[];
  __bf16* sA = (__bf16*)smem;               // MT x NDIM, swizzled
  __bf16* sH = (__bf16*)(smem + SMEM_A);    // MT x HDIM, swizzled

  const int e    = blockIdx.y;
  const int tile = blockIdx.x;
  const int cnt  = counts[e];
  if (tile * MT >= cnt) return;

  const int tid = threadIdx.x;

  { // ---- stage x tile: 32 rows, 8 threads/row, 16 chunks of 16B each ----
    int row = tid >> 3;
    int sub = tid & 7;
    int gidx = tile * MT + row;
    if (gidx < cnt) {
      int tok = perm[e * NTOK + gidx];
      const __bf16* src = xb + (size_t)tok * NDIM;
      #pragma unroll
      for (int c = sub * 16; c < sub * 16 + 16; ++c) {
        unsigned ldsOff = (unsigned)(SMEM_A * 0 + swz_off(row, c << 3, NDIM) * 2);
        async_copy16(ldsOff, src + (c << 3));
      }
    } else {
      v8bf z;
      #pragma unroll
      for (int i = 0; i < 8; ++i) z[i] = (__bf16)0.f;
      #pragma unroll
      for (int c = sub * 16; c < sub * 16 + 16; ++c)
        *(v8bf*)(sA + swz_off(row, c << 3, NDIM)) = z;
    }
    async_wait0();
  }
  __syncthreads();

  const int lane = tid & 31;
  const int wv   = tid >> 5;
  const int col  = lane & 15;               // A row within sub-tile / B,C col
  const int kb   = (lane < 16) ? 0 : 8;     // K sub-block per ISA layout
  const int m0   = (lane < 16) ? 0 : 8;     // C/D M base within sub-tile

  // ---- GEMM1: h = relu(x @ W1[e] + b1[e]) ----
  for (int nt = wv; nt < HDIM / 16; nt += 8) {
    const int n0 = nt * 16;
    v8f acc0 = {0.f,0.f,0.f,0.f,0.f,0.f,0.f,0.f};
    v8f acc1 = {0.f,0.f,0.f,0.f,0.f,0.f,0.f,0.f};
    const __bf16* bbase = W1t + ((size_t)e * HDIM + n0 + col) * NDIM + kb;
    #pragma unroll 2
    for (int k0 = 0; k0 < NDIM; k0 += 32) {
      v16bf b  = load_frag(bbase + k0);
      __builtin_prefetch(bbase + k0 + 256, 0, 1);
      v16bf a0 = load_frag_swz(sA, col,      k0 + kb, NDIM);
      v16bf a1 = load_frag_swz(sA, 16 + col, k0 + kb, NDIM);
      acc0 = __builtin_amdgcn_wmma_f32_16x16x32_bf16(false, a0, false, b,
                                                     (short)0, acc0, false, false);
      acc1 = __builtin_amdgcn_wmma_f32_16x16x32_bf16(false, a1, false, b,
                                                     (short)0, acc1, false, false);
    }
    const float bias = b1[e * HDIM + n0 + col];
    #pragma unroll
    for (int r = 0; r < 8; ++r) {
      sH[swz_off(m0 + r,      n0 + col, HDIM)] = (__bf16)fmaxf(acc0[r] + bias, 0.f);
      sH[swz_off(16 + m0 + r, n0 + col, HDIM)] = (__bf16)fmaxf(acc1[r] + bias, 0.f);
    }
  }
  __syncthreads();

  // ---- GEMM2: y += cw * (h @ W2[e] + b2[e]) ----
  for (int nt = wv; nt < NDIM / 16; nt += 8) {
    const int n0 = nt * 16;
    v8f acc0 = {0.f,0.f,0.f,0.f,0.f,0.f,0.f,0.f};
    v8f acc1 = {0.f,0.f,0.f,0.f,0.f,0.f,0.f,0.f};
    const __bf16* bbase = W2t + ((size_t)e * NDIM + n0 + col) * HDIM + kb;
    #pragma unroll 2
    for (int k0 = 0; k0 < HDIM; k0 += 32) {
      v16bf b  = load_frag(bbase + k0);
      __builtin_prefetch(bbase + k0 + 256, 0, 1);
      v16bf a0 = load_frag_swz(sH, col,      k0 + kb, HDIM);
      v16bf a1 = load_frag_swz(sH, 16 + col, k0 + kb, HDIM);
      acc0 = __builtin_amdgcn_wmma_f32_16x16x32_bf16(false, a0, false, b,
                                                     (short)0, acc0, false, false);
      acc1 = __builtin_amdgcn_wmma_f32_16x16x32_bf16(false, a1, false, b,
                                                     (short)0, acc1, false, false);
    }
    const float bias = b2[e * NDIM + n0 + col];
    #pragma unroll
    for (int s = 0; s < 2; ++s) {
      const v8f& a = s ? acc1 : acc0;
      #pragma unroll
      for (int r = 0; r < 8; ++r) {
        int m = s * 16 + m0 + r;
        int gidx = tile * MT + m;
        if (gidx < cnt) {
          int tok = perm[e * NTOK + gidx];
          float w = cwt[e * NTOK + gidx];
          atomicAdd(y + (size_t)tok * NDIM + n0 + col, w * (a[r] + bias));
        }
      }
    }
  }
}

// ---------------------------------------------------------------------------
extern "C" void kernel_launch(void* const* d_in, const int* in_sizes, int n_in,
                              void* d_out, int out_size, void* d_ws, size_t ws_size,
                              hipStream_t stream) {
  (void)in_sizes; (void)n_in; (void)out_size; (void)ws_size;
  const float* x  = (const float*)d_in[0];
  const float* Wg = (const float*)d_in[1];
  const float* bg = (const float*)d_in[2];
  const float* W1 = (const float*)d_in[3];
  const float* b1 = (const float*)d_in[4];
  const float* W2 = (const float*)d_in[5];
  const float* b2 = (const float*)d_in[6];

  float* out = (float*)d_out;
  float* y   = out;                                      // [B, D]
  float* aux = out + (size_t)NTOK * NDIM;                // [1]
  float* gp  = aux + 1;                                  // [B, E]
  float* hm  = gp + (size_t)NTOK * NEXP;                 // [B, E]

  char* ws = (char*)d_ws;
  size_t off = 0;
  __bf16* xb  = (__bf16*)(ws + off); off += (size_t)NTOK * NDIM * 2;         // 16 MB
  __bf16* W1t = (__bf16*)(ws + off); off += (size_t)NEXP * HDIM * NDIM * 2;  // 64 MB
  __bf16* W2t = (__bf16*)(ws + off); off += (size_t)NEXP * NDIM * HDIM * 2;  // 64 MB
  int*   counts    = (int*)  (ws + off); off += 256;
  float* sum_probs = (float*)(ws + off); off += 256;
  int*   perm      = (int*)  (ws + off); off += (size_t)NEXP * NTOK * 4;
  float* cwt       = (float*)(ws + off);

  hipMemsetAsync(y, 0, (size_t)NTOK * NDIM * sizeof(float), stream);
  hipMemsetAsync(counts, 0, 512, stream);   // counts + sum_probs

  moe_cvt_x_kernel <<<4096, 256, 0, stream>>>(x,  xb);
  moe_cvt_w1_kernel<<<8192, 256, 0, stream>>>(W1, W1t);
  moe_cvt_w2_kernel<<<8192, 256, 0, stream>>>(W2, W2t);

  moe_gate_kernel<<<NTOK / 8, 256, 0, stream>>>(x, Wg, bg, gp, hm,
                                                sum_probs, counts, perm, cwt);
  moe_aux_kernel<<<1, 32, 0, stream>>>(sum_probs, aux);

  dim3 grid(NTOK / MT, NEXP);
  moe_ffn_kernel<<<grid, 256, SMEM_BYTES, stream>>>(xb, W1t, b1, W2t, b2,
                                                    counts, perm, cwt, y);
}
